// InteractionNetwork_80066780332694
// MI455X (gfx1250) — compile-verified, matrix operand order checked by reading, not run on hardware
//
#include <hip/hip_runtime.h>
#include <hip/hip_bf16.h>

typedef __bf16 v8bf  __attribute__((ext_vector_type(8)));
typedef __bf16 v16bf __attribute__((ext_vector_type(16)));
typedef float  v8f   __attribute__((ext_vector_type(8)));

__device__ __forceinline__ v16bf cat16(const __bf16* p0, const __bf16* p1) {
    v8bf lo = *(const v8bf*)p0;
    v8bf hi = *(const v8bf*)p1;
    return __builtin_shufflevector(lo, hi, 0,1,2,3,4,5,6,7,8,9,10,11,12,13,14,15);
}

__device__ __forceinline__ v8f wmma_bf16(v16bf a, v16bf b, v8f c) {
    return __builtin_amdgcn_wmma_f32_16x16x32_bf16(false, a, false, b, (short)0, c, false, false);
}

// ---------- weight transpose + bf16 convert: W[K,N] -> WT[N,K] ----------
__global__ void wt_bf16_kernel(const float* __restrict__ w, __bf16* __restrict__ wt,
                               int K, int N) {
    int idx = blockIdx.x * blockDim.x + threadIdx.x;
    if (idx < K * N) {
        int k = idx / N, n = idx % N;
        wt[(long)n * K + k] = (__bf16)w[idx];
    }
}

// ---------- encoders (tiny fan-in, scalar f32) ----------
__global__ __launch_bounds__(64)
void encode_nodes_kernel(const float* __restrict__ x,
                         const float* __restrict__ w1, const float* __restrict__ b1,
                         const float* __restrict__ w2, const float* __restrict__ b2,
                         __bf16* __restrict__ x_out, __bf16* __restrict__ save_out, int n_nodes) {
    __shared__ float sh[64];
    long n = blockIdx.x; int j = threadIdx.x;
    if (n >= n_nodes) return;
    float a = b1[j];
    #pragma unroll
    for (int k = 0; k < 3; ++k) a += x[n * 3 + k] * w1[k * 64 + j];
    sh[j] = a > 0.f ? a : 0.f;
    __syncthreads();
    float o = b2[j];
    #pragma unroll 8
    for (int k = 0; k < 64; ++k) o += sh[k] * w2[k * 64 + j];
    o = o > 0.f ? o : 0.f;
    __bf16 v = (__bf16)o;
    x_out[n * 64 + j] = v;
    save_out[n * 64 + j] = v;
}

__global__ __launch_bounds__(64)
void encode_edges_kernel(const float* __restrict__ ea,
                         const float* __restrict__ w1, const float* __restrict__ b1,
                         const float* __restrict__ w2, const float* __restrict__ b2,
                         __bf16* __restrict__ e_out, __bf16* __restrict__ save_out, int n_edges) {
    __shared__ float sh[64];
    long e = blockIdx.x; int j = threadIdx.x;
    if (e >= n_edges) return;
    float a = b1[j];
    #pragma unroll
    for (int k = 0; k < 2; ++k) a += ea[e * 2 + k] * w1[k * 64 + j];
    sh[j] = a > 0.f ? a : 0.f;
    __syncthreads();
    float o = b2[j];
    #pragma unroll 8
    for (int k = 0; k < 64; ++k) o += sh[k] * w2[k * 64 + j];
    o = o > 0.f ? o : 0.f;
    __bf16 v = (__bf16)o;
    e_out[e * 64 + j] = v;
    save_out[e * 64 + j] = v;
}

// ---------- fused per-edge layer: edge MLP + node message MLP + scatter ----------
__global__ __launch_bounds__(128)
void edge_layer_kernel(const __bf16* __restrict__ x_cur, const __bf16* __restrict__ save_x,
                       const __bf16* __restrict__ e_cur, const __bf16* __restrict__ save_e,
                       const int* __restrict__ senders, const int* __restrict__ receivers,
                       const __bf16* __restrict__ wt_e1, const float* __restrict__ b_e1,
                       const __bf16* __restrict__ wt_e2, const float* __restrict__ b_e2,
                       const __bf16* __restrict__ wt_n1a, const float* __restrict__ b_n1a,
                       const __bf16* __restrict__ wt_n1b, const float* __restrict__ b_n1b,
                       __bf16* __restrict__ e_new, float* __restrict__ agg, int n_edges) {
    __shared__ __align__(16) __bf16 sA[16][392];  // cat(xc[s],xc[r],ec) : 16 edges x 384 (+pad)
    __shared__ __align__(16) __bf16 sH[16][72];   // hidden 64 (+pad)
    __shared__ __align__(16) __bf16 sE[16][64];   // new edge features
    __shared__ int sS[16], sR[16];

    const int tid  = threadIdx.x;
    const int wave = tid >> 5, lane = tid & 31;
    const int h = lane >> 4, m = lane & 15;
    const int colg = wave * 16 + (lane & 15);
    const long e0 = (long)blockIdx.x * 16;

    if (tid < 16) {
        long ei = e0 + tid; if (ei >= n_edges) ei = n_edges - 1;
        sS[tid] = senders[ei];
        sR[tid] = receivers[ei];
    }
    __syncthreads();

    // stage gathered A (bf16, 16B chunks): 16 edges x 48 chunks of 8
    for (int c = tid; c < 16 * 48; c += 128) {
        int i = c / 48, f = (c % 48) * 8;
        int seg = f >> 6, offn = f & 63;
        long ei = e0 + i; if (ei >= n_edges) ei = n_edges - 1;
        const __bf16* src;
        switch (seg) {
            case 0:  src = x_cur  + (long)sS[i] * 64 + offn; break;
            case 1:  src = save_x + (long)sS[i] * 64 + offn; break;
            case 2:  src = x_cur  + (long)sR[i] * 64 + offn; break;
            case 3:  src = save_x + (long)sR[i] * 64 + offn; break;
            case 4:  src = e_cur  + ei * 64 + offn;          break;
            default: src = save_e + ei * 64 + offn;          break;
        }
        *(uint4*)&sA[i][f] = *(const uint4*)src;
    }
    __syncthreads();

    // ---- edge_mlp1: [16x384] @ [384x64], K = 12 x 32 ----
    v8f acc = {};
    {
        const __bf16* wrow = wt_e1 + (long)colg * 384;
        #pragma unroll
        for (int kk = 0; kk < 12; ++kk) {
            int k0 = kk * 32;
            v16bf a = cat16(&sA[m][k0 + h * 8], &sA[m][k0 + 16 + h * 8]);
            v16bf b = cat16(wrow + k0 + h * 8, wrow + k0 + 16 + h * 8);
            acc = wmma_bf16(a, b, acc);
        }
        float bias = b_e1[colg];
        #pragma unroll
        for (int r = 0; r < 8; ++r) {
            float v = acc[r] + bias; v = v > 0.f ? v : 0.f;
            sH[r + h * 8][colg] = (__bf16)v;
        }
    }
    __syncthreads();

    // ---- edge_mlp2: 64 -> 64 ----
    acc = v8f{};
    {
        const __bf16* wrow = wt_e2 + (long)colg * 64;
        #pragma unroll
        for (int kk = 0; kk < 2; ++kk) {
            int k0 = kk * 32;
            v16bf a = cat16(&sH[m][k0 + h * 8], &sH[m][k0 + 16 + h * 8]);
            v16bf b = cat16(wrow + k0 + h * 8, wrow + k0 + 16 + h * 8);
            acc = wmma_bf16(a, b, acc);
        }
        float bias = b_e2[colg];
        #pragma unroll
        for (int r = 0; r < 8; ++r) {
            float v = acc[r] + bias; v = v > 0.f ? v : 0.f;
            sE[r + h * 8][colg] = (__bf16)v;
        }
    }
    __syncthreads();

    // coalesced store of new edge features
    {
        int row = tid >> 3, c0 = (tid & 7) * 8;
        long ei = e0 + row;
        if (ei < n_edges)
            *(uint4*)(e_new + ei * 64 + c0) = *(const uint4*)&sE[row][c0];
    }

    // ---- node_mlp1a: cat(xc[s](128), e_new(64)) @ [192x64] ----
    acc = v8f{};
    {
        const __bf16* wrow = wt_n1a + (long)colg * 192;
        #pragma unroll
        for (int kk = 0; kk < 6; ++kk) {
            int k0 = kk * 32;
            const __bf16 *pa0, *pa1;
            if (k0 < 128) { pa0 = &sA[m][k0 + h * 8];       pa1 = &sA[m][k0 + 16 + h * 8]; }
            else          { pa0 = &sE[m][k0 - 128 + h * 8]; pa1 = &sE[m][k0 - 128 + 16 + h * 8]; }
            v16bf a = cat16(pa0, pa1);
            v16bf b = cat16(wrow + k0 + h * 8, wrow + k0 + 16 + h * 8);
            acc = wmma_bf16(a, b, acc);
        }
        float bias = b_n1a[colg];
        #pragma unroll
        for (int r = 0; r < 8; ++r) {
            float v = acc[r] + bias; v = v > 0.f ? v : 0.f;
            sH[r + h * 8][colg] = (__bf16)v;
        }
    }
    __syncthreads();

    // ---- node_mlp1b: 64 -> 64, then scatter-add to agg[receivers] ----
    acc = v8f{};
    {
        const __bf16* wrow = wt_n1b + (long)colg * 64;
        #pragma unroll
        for (int kk = 0; kk < 2; ++kk) {
            int k0 = kk * 32;
            v16bf a = cat16(&sH[m][k0 + h * 8], &sH[m][k0 + 16 + h * 8]);
            v16bf b = cat16(wrow + k0 + h * 8, wrow + k0 + 16 + h * 8);
            acc = wmma_bf16(a, b, acc);
        }
        float bias = b_n1b[colg];
        #pragma unroll
        for (int r = 0; r < 8; ++r) {
            float v = acc[r] + bias; v = v > 0.f ? v : 0.f;
            int mr = r + h * 8;
            long ei = e0 + mr;
            if (ei < n_edges)
                atomicAdd(&agg[(long)sR[mr] * 64 + colg], v);
        }
    }
}

// ---------- node update: x = relu(mlp2b(relu(mlp2a(cat(x, save_x, agg))))) ----------
__global__ __launch_bounds__(128)
void node_update_kernel(const __bf16* __restrict__ x_cur, const __bf16* __restrict__ save_x,
                        const float* __restrict__ agg,
                        const __bf16* __restrict__ wt_a, const float* __restrict__ b_a,
                        const __bf16* __restrict__ wt_b, const float* __restrict__ b_b,
                        __bf16* __restrict__ x_new, int n_nodes) {
    __shared__ __align__(16) __bf16 sA[16][200];  // 192 (+pad)
    __shared__ __align__(16) __bf16 sH[16][72];
    __shared__ __align__(16) __bf16 sE[16][64];

    const int tid  = threadIdx.x;
    const int wave = tid >> 5, lane = tid & 31;
    const int h = lane >> 4, m = lane & 15;
    const int colg = wave * 16 + (lane & 15);
    const long n0 = (long)blockIdx.x * 16;

    // stage cat(x, save_x, agg): 16 nodes x 24 chunks of 8
    for (int c = tid; c < 16 * 24; c += 128) {
        int i = c / 24, f = (c % 24) * 8;
        int seg = f >> 6, offn = f & 63;
        long ni = n0 + i; if (ni >= n_nodes) ni = n_nodes - 1;
        if (seg == 0) {
            *(uint4*)&sA[i][f] = *(const uint4*)(x_cur + ni * 64 + offn);
        } else if (seg == 1) {
            *(uint4*)&sA[i][f] = *(const uint4*)(save_x + ni * 64 + offn);
        } else {
            const float* fs = agg + ni * 64 + offn;
            __bf16* dst = &sA[i][f];
            #pragma unroll
            for (int q = 0; q < 8; ++q) dst[q] = (__bf16)fs[q];
        }
    }
    __syncthreads();

    v8f acc = {};
    {
        const __bf16* wrow = wt_a + (long)colg * 192;
        #pragma unroll
        for (int kk = 0; kk < 6; ++kk) {
            int k0 = kk * 32;
            v16bf a = cat16(&sA[m][k0 + h * 8], &sA[m][k0 + 16 + h * 8]);
            v16bf b = cat16(wrow + k0 + h * 8, wrow + k0 + 16 + h * 8);
            acc = wmma_bf16(a, b, acc);
        }
        float bias = b_a[colg];
        #pragma unroll
        for (int r = 0; r < 8; ++r) {
            float v = acc[r] + bias; v = v > 0.f ? v : 0.f;
            sH[r + h * 8][colg] = (__bf16)v;
        }
    }
    __syncthreads();

    acc = v8f{};
    {
        const __bf16* wrow = wt_b + (long)colg * 64;
        #pragma unroll
        for (int kk = 0; kk < 2; ++kk) {
            int k0 = kk * 32;
            v16bf a = cat16(&sH[m][k0 + h * 8], &sH[m][k0 + 16 + h * 8]);
            v16bf b = cat16(wrow + k0 + h * 8, wrow + k0 + 16 + h * 8);
            acc = wmma_bf16(a, b, acc);
        }
        float bias = b_b[colg];
        #pragma unroll
        for (int r = 0; r < 8; ++r) {
            float v = acc[r] + bias; v = v > 0.f ? v : 0.f;
            sE[r + h * 8][colg] = (__bf16)v;
        }
    }
    __syncthreads();

    {
        int row = tid >> 3, c0 = (tid & 7) * 8;
        long ni = n0 + row;
        if (ni < n_nodes)
            *(uint4*)(x_new + ni * 64 + c0) = *(const uint4*)&sE[row][c0];
    }
}

// ---------- decoder: 64 -> 256 -> 256 -> 1 (scalar f32) ----------
__global__ __launch_bounds__(256)
void decoder_kernel(const __bf16* __restrict__ xin,
                    const float* __restrict__ w3, const float* __restrict__ b3,
                    const float* __restrict__ w4, const float* __restrict__ b4,
                    const float* __restrict__ w5, const float* __restrict__ b5,
                    float* __restrict__ out, int n_nodes) {
    __shared__ float sx[64];
    __shared__ float sh1[256];
    __shared__ float sred[256];
    long n = blockIdx.x; int j = threadIdx.x;
    if (n >= n_nodes) return;
    if (j < 64) sx[j] = (float)xin[n * 64 + j];
    __syncthreads();
    float h1 = b3[j];
    #pragma unroll 8
    for (int k = 0; k < 64; ++k) h1 += sx[k] * w3[k * 256 + j];
    sh1[j] = h1 > 0.f ? h1 : 0.f;
    __syncthreads();
    float h2 = b4[j];
    #pragma unroll 8
    for (int k = 0; k < 256; ++k) h2 += sh1[k] * w4[k * 256 + j];
    h2 = h2 > 0.f ? h2 : 0.f;
    sred[j] = h2 * w5[j];
    __syncthreads();
    for (int s = 128; s > 0; s >>= 1) {
        if (j < s) sred[j] += sred[j + s];
        __syncthreads();
    }
    if (j == 0) out[n] = sred[0] + b5[0];
}

extern "C" void kernel_launch(void* const* d_in, const int* in_sizes, int n_in,
                              void* d_out, int out_size, void* d_ws, size_t ws_size,
                              hipStream_t stream) {
    const float* x   = (const float*)d_in[0];
    const float* ea  = (const float*)d_in[1];
    const int* eidx  = (const int*)d_in[2];
    const int n_nodes = in_sizes[0] / 3;
    const int n_edges = in_sizes[1] / 2;
    const int* senders   = eidx;
    const int* receivers = eidx + n_edges;

    // params flattened in dict order, (w,b) pairs starting at d_in[3]
    const float* w_fc1 = (const float*)d_in[3];   const float* b_fc1 = (const float*)d_in[4];
    const float* w_fc2 = (const float*)d_in[5];   const float* b_fc2 = (const float*)d_in[6];
    const float* w_fc3 = (const float*)d_in[7];   const float* b_fc3 = (const float*)d_in[8];
    const float* w_fc4 = (const float*)d_in[9];   const float* b_fc4 = (const float*)d_in[10];
    const float* w_fc5 = (const float*)d_in[11];  const float* b_fc5 = (const float*)d_in[12];
    const float* w_ef1 = (const float*)d_in[13];  const float* b_ef1 = (const float*)d_in[14];
    const float* w_ef2 = (const float*)d_in[15];  const float* b_ef2 = (const float*)d_in[16];
    const float* w_e1  = (const float*)d_in[17];  const float* b_e1  = (const float*)d_in[18];
    const float* w_e2  = (const float*)d_in[19];  const float* b_e2  = (const float*)d_in[20];
    const float* w_n1a = (const float*)d_in[21];  const float* b_n1a = (const float*)d_in[22];
    const float* w_n1b = (const float*)d_in[23];  const float* b_n1b = (const float*)d_in[24];
    const float* w_n2a = (const float*)d_in[25];  const float* b_n2a = (const float*)d_in[26];
    const float* w_n2b = (const float*)d_in[27];  const float* b_n2b = (const float*)d_in[28];

    char* ws = (char*)d_ws;
    size_t off = 0;
    auto take = [&](size_t bytes) -> char* {
        char* p = ws + off;
        off = (off + bytes + 255) & ~(size_t)255;
        return p;
    };

    __bf16* wt_e1  = (__bf16*)take((size_t)64 * 384 * 2);
    __bf16* wt_e2  = (__bf16*)take((size_t)64 * 64 * 2);
    __bf16* wt_n1a = (__bf16*)take((size_t)64 * 192 * 2);
    __bf16* wt_n1b = (__bf16*)take((size_t)64 * 64 * 2);
    __bf16* wt_n2a = (__bf16*)take((size_t)64 * 192 * 2);
    __bf16* wt_n2b = (__bf16*)take((size_t)64 * 64 * 2);
    __bf16* x_a    = (__bf16*)take((size_t)n_nodes * 64 * 2);
    __bf16* x_b    = (__bf16*)take((size_t)n_nodes * 64 * 2);
    __bf16* save_x = (__bf16*)take((size_t)n_nodes * 64 * 2);
    __bf16* e_a    = (__bf16*)take((size_t)n_edges * 64 * 2);
    __bf16* e_b    = (__bf16*)take((size_t)n_edges * 64 * 2);
    __bf16* save_e = (__bf16*)take((size_t)n_edges * 64 * 2);
    float*  agg    = (float*) take((size_t)n_nodes * 64 * 4);
    (void)ws_size; (void)n_in; (void)out_size;

    auto wt = [&](const float* w, __bf16* dst, int K, int N) {
        int total = K * N;
        wt_bf16_kernel<<<(total + 255) / 256, 256, 0, stream>>>(w, dst, K, N);
    };
    wt(w_e1,  wt_e1,  384, 64);
    wt(w_e2,  wt_e2,  64,  64);
    wt(w_n1a, wt_n1a, 192, 64);
    wt(w_n1b, wt_n1b, 64,  64);
    wt(w_n2a, wt_n2a, 192, 64);
    wt(w_n2b, wt_n2b, 64,  64);

    encode_nodes_kernel<<<n_nodes, 64, 0, stream>>>(x, w_fc1, b_fc1, w_fc2, b_fc2,
                                                    x_a, save_x, n_nodes);
    encode_edges_kernel<<<n_edges, 64, 0, stream>>>(ea, w_ef1, b_ef1, w_ef2, b_ef2,
                                                    e_a, save_e, n_edges);

    __bf16 *xc = x_a, *xn = x_b, *ec = e_a, *en = e_b;
    const int gridE = (n_edges + 15) / 16;
    const int gridN = (n_nodes + 15) / 16;
    for (int l = 0; l < 7; ++l) {
        hipMemsetAsync(agg, 0, (size_t)n_nodes * 64 * sizeof(float), stream);
        edge_layer_kernel<<<gridE, 128, 0, stream>>>(
            xc, save_x, ec, save_e, senders, receivers,
            wt_e1, b_e1, wt_e2, b_e2, wt_n1a, b_n1a, wt_n1b, b_n1b,
            en, agg, n_edges);
        node_update_kernel<<<gridN, 128, 0, stream>>>(
            xc, save_x, agg, wt_n2a, b_n2a, wt_n2b, b_n2b, xn, n_nodes);
        __bf16* t;
        t = xc; xc = xn; xn = t;
        t = ec; ec = en; en = t;
    }

    decoder_kernel<<<n_nodes, 256, 0, stream>>>(xc, w_fc3, b_fc3, w_fc4, b_fc4,
                                                w_fc5, b_fc5, (float*)d_out, n_nodes);
}
